// MultiHeadAttention_2241972929037
// MI455X (gfx1250) — matrix operand order repaired and checked by reading, not compiled, not run on hardware
//
#include <hip/hip_runtime.h>
#include <hip/hip_bf16.h>

// CDNA5 / gfx1250, wave32. WMMA 16x16x32 bf16 + async Global->LDS staging.

typedef __attribute__((ext_vector_type(16))) __bf16 v16bf;
typedef __attribute__((ext_vector_type(8)))  float  v8f;

union Frag32 { uint4 u[2]; v16bf v; };   // 32 bytes = 16 bf16

#define WMMA_BF16(a, b, c) \
  __builtin_amdgcn_wmma_f32_16x16x32_bf16(false, (a), false, (b), (short)0, (c), false, false)

__device__ __forceinline__ unsigned short f2bf(float f) {
  // round-to-nearest-even fp32 -> bf16 (bit trick, no __bf16 arithmetic)
  unsigned int u = __float_as_uint(f);
  unsigned int r = u + 0x7FFFu + ((u >> 16) & 1u);
  return (unsigned short)(r >> 16);
}

// async DMA: 16 bytes global -> LDS, tracked by ASYNCcnt (CDNA5)
__device__ __forceinline__ void async_copy_b128(unsigned lds_byte,
                                                const unsigned short* g) {
  asm volatile("global_load_async_to_lds_b128 %0, %1, off"
               :: "v"(lds_byte), "v"((unsigned long long)(size_t)g)
               : "memory");
}

// ---------------------------------------------------------------- convert
__global__ void cvt_f32_bf16(const float* __restrict__ in,
                             unsigned short* __restrict__ out, int n) {
  int i = blockIdx.x * blockDim.x + threadIdx.x;
  int stride = gridDim.x * blockDim.x;
  for (; i < n; i += stride) out[i] = f2bf(in[i]);
}

// ---------------------------------------------------------------- QKV projection
// Y = X @ W^T + b; X:(4096,1024) bf16, W:(1024,1024) bf16, row-major.
// One 32x32 output tile per wave (2 A-frags x 2 B-frags -> 4 WMMA / k-step).
// vmode 0: head-split (b,h,s,64) store (Q,K). vmode 1: transposed (b,h,64,s) store (V).
__global__ void proj32(const unsigned short* __restrict__ X,
                       const unsigned short* __restrict__ W,
                       const float* __restrict__ bias,
                       unsigned short* __restrict__ Y,
                       int vmode) {
  int wave = threadIdx.x >> 5;
  int lane = threadIdx.x & 31;
  int w = blockIdx.x * (blockDim.x >> 5) + wave;
  int mt = w >> 5, nt = w & 31;            // 128 x 32 tiles of 32x32
  int i0 = mt * 32, j0 = nt * 32;
  int hi = lane >> 4, lo = lane & 15;

  const unsigned short* a0p = X + (size_t)(i0 + lo) * 1024 + 8 * hi;
  const unsigned short* a1p = a0p + (size_t)16 * 1024;
  const unsigned short* b0p = W + (size_t)(j0 + lo) * 1024 + 8 * hi;
  const unsigned short* b1p = b0p + (size_t)16 * 1024;

  v8f acc00 = {}, acc01 = {}, acc10 = {}, acc11 = {};
  for (int kk = 0; kk < 1024; kk += 32) {
    Frag32 a0, a1, b0, b1;
    a0.u[0] = *(const uint4*)(a0p + kk);     a0.u[1] = *(const uint4*)(a0p + kk + 16);
    a1.u[0] = *(const uint4*)(a1p + kk);     a1.u[1] = *(const uint4*)(a1p + kk + 16);
    b0.u[0] = *(const uint4*)(b0p + kk);     b0.u[1] = *(const uint4*)(b0p + kk + 16);
    b1.u[0] = *(const uint4*)(b1p + kk);     b1.u[1] = *(const uint4*)(b1p + kk + 16);
    acc00 = WMMA_BF16(a0.v, b0.v, acc00);
    acc01 = WMMA_BF16(a0.v, b1.v, acc01);
    acc10 = WMMA_BF16(a1.v, b0.v, acc10);
    acc11 = WMMA_BF16(a1.v, b1.v, acc11);
  }

  for (int ci = 0; ci < 2; ++ci) {
    int j = j0 + 16 * ci + lo;
    float bj = bias[j];
    int h = j >> 6, d = j & 63;
    for (int ri = 0; ri < 2; ++ri) {
      const v8f& acc = (ri == 0) ? (ci == 0 ? acc00 : acc01)
                                 : (ci == 0 ? acc10 : acc11);
      for (int r = 0; r < 8; ++r) {          // C/D layout: row = r + 8*hi, col = lo
        int i = i0 + 16 * ri + r + 8 * hi;
        int bb = i >> 11, s = i & 2047;
        size_t addr = vmode
            ? ((((size_t)bb * 16 + h) * 64 + d) * 2048 + s)     // (b,h,d,s)
            : ((((size_t)bb * 16 + h) * 2048 + s) * 64 + d);    // (b,h,s,d)
        Y[addr] = f2bf(acc[r] + bj);
      }
    }
  }
}

// ---------------------------------------------------------------- output projection
__global__ void proj_out32(const unsigned short* __restrict__ X,
                           const unsigned short* __restrict__ W,
                           const float* __restrict__ bias,
                           float* __restrict__ Y) {
  int wave = threadIdx.x >> 5;
  int lane = threadIdx.x & 31;
  int w = blockIdx.x * (blockDim.x >> 5) + wave;
  int mt = w >> 5, nt = w & 31;
  int i0 = mt * 32, j0 = nt * 32;
  int hi = lane >> 4, lo = lane & 15;

  const unsigned short* a0p = X + (size_t)(i0 + lo) * 1024 + 8 * hi;
  const unsigned short* a1p = a0p + (size_t)16 * 1024;
  const unsigned short* b0p = W + (size_t)(j0 + lo) * 1024 + 8 * hi;
  const unsigned short* b1p = b0p + (size_t)16 * 1024;

  v8f acc00 = {}, acc01 = {}, acc10 = {}, acc11 = {};
  for (int kk = 0; kk < 1024; kk += 32) {
    Frag32 a0, a1, b0, b1;
    a0.u[0] = *(const uint4*)(a0p + kk);     a0.u[1] = *(const uint4*)(a0p + kk + 16);
    a1.u[0] = *(const uint4*)(a1p + kk);     a1.u[1] = *(const uint4*)(a1p + kk + 16);
    b0.u[0] = *(const uint4*)(b0p + kk);     b0.u[1] = *(const uint4*)(b0p + kk + 16);
    b1.u[0] = *(const uint4*)(b1p + kk);     b1.u[1] = *(const uint4*)(b1p + kk + 16);
    acc00 = WMMA_BF16(a0.v, b0.v, acc00);
    acc01 = WMMA_BF16(a0.v, b1.v, acc01);
    acc10 = WMMA_BF16(a1.v, b0.v, acc10);
    acc11 = WMMA_BF16(a1.v, b1.v, acc11);
  }

  for (int ci = 0; ci < 2; ++ci) {
    int j = j0 + 16 * ci + lo;
    float bj = bias[j];
    for (int ri = 0; ri < 2; ++ri) {
      const v8f& acc = (ri == 0) ? (ci == 0 ? acc00 : acc01)
                                 : (ci == 0 ? acc10 : acc11);
      for (int r = 0; r < 8; ++r) {
        int i = i0 + 16 * ri + r + 8 * hi;
        Y[(size_t)i * 1024 + j] = acc[r] + bj;
      }
    }
  }
}

// ---------------------------------------------------------------- flash attention
// 4 waves/block, all sharing one (b,h); each wave owns a 16-row Q tile.
// Double-buffered ASYNC DMA staging: while computing key-tile i from LDS,
// tile i+1 streams in via global_load_async_to_lds_b128 (ASYNCcnt).
__global__ void flash64(const unsigned short* __restrict__ Qh,
                        const unsigned short* __restrict__ Kh,
                        const unsigned short* __restrict__ Vt,
                        const int* __restrict__ mask,
                        unsigned short* __restrict__ Ob) {
  __shared__ uint4 sK4[2][256];                    // 2 x (32 x 64 bf16) key-major
  __shared__ uint4 sV4[2][256];                    // 2 x (64 x 32 bf16) d-major
  __shared__ uint4 sP4[256];                       // 4 waves x 16x32 bf16
  unsigned short* sP = (unsigned short*)sP4;

  int tid = threadIdx.x;
  int wave = tid >> 5, lane = tid & 31;
  int w = blockIdx.x * 4 + wave;
  int bh = w >> 7;                                 // uniform across block
  int qt = w & 127;
  int b = bh >> 4, h = bh & 15;
  int hi = lane >> 4, lo = lane & 15;

  const unsigned short* Qb = Qh + (size_t)bh * 2048 * 64;
  const unsigned short* Kb = Kh + (size_t)bh * 2048 * 64;
  const unsigned short* Vb = Vt + (size_t)bh * 64 * 2048;
  const int* mg = mask + b * 2048;
  int q0 = qt * 16;

  // Q as two A-frags (d=0..31, d=32..63), held in VGPRs for the whole pass
  Frag32 qa0, qa1;
  {
    const unsigned short* qp = Qb + (size_t)(q0 + lo) * 64 + 8 * hi;
    qa0.u[0] = *(const uint4*)(qp);
    qa0.u[1] = *(const uint4*)(qp + 16);
    qa1.u[0] = *(const uint4*)(qp + 32);
    qa1.u[1] = *(const uint4*)(qp + 48);
  }

  float mr[8], lr[8];
  for (int r = 0; r < 8; ++r) { mr[r] = -1e30f; lr[r] = 0.0f; }
  v8f o0 = {}, o1 = {}, o2 = {}, o3 = {};
  unsigned short* myp = sP + wave * 512;

  // per-thread DMA assignments (128 threads)
  int vrow = tid >> 1;                  // Vt d-row 0..63
  int vhalf = (tid & 1) * 16;           // 16-ushort half of a 32-key row
  unsigned kldsA = (unsigned)(size_t)&sK4[0][0] + (unsigned)tid * 32u;
  unsigned kldsB = (unsigned)(size_t)&sK4[1][0] + (unsigned)tid * 32u;
  unsigned vldsA = (unsigned)(size_t)&sV4[0][0] + (unsigned)(vrow * 32 + vhalf) * 2u;
  unsigned vldsB = (unsigned)(size_t)&sV4[1][0] + (unsigned)(vrow * 32 + vhalf) * 2u;

  // prologue: start DMA for tile 0 into buffer 0
  {
    const unsigned short* ks = Kb + (size_t)tid * 16;        // j32 = 0
    const unsigned short* vs = Vb + (size_t)vrow * 2048 + vhalf;
    async_copy_b128(kldsA,      ks);
    async_copy_b128(kldsA + 16, ks + 8);
    async_copy_b128(vldsA,      vs);
    async_copy_b128(vldsA + 16, vs + 8);
  }

  for (int it = 0; it < 64; ++it) {
    int j32 = it * 32;
    // issue DMA for next tile into the other buffer (free since it-1's compute
    // finished behind two barriers), then wait for THIS tile's 4 async ops.
    if (it + 1 < 64) {
      int jn = j32 + 32;
      unsigned kl = ((it + 1) & 1) ? kldsB : kldsA;
      unsigned vl = ((it + 1) & 1) ? vldsB : vldsA;
      const unsigned short* ks = Kb + (size_t)jn * 64 + (size_t)tid * 16;
      const unsigned short* vs = Vb + (size_t)vrow * 2048 + jn + vhalf;
      async_copy_b128(kl,      ks);
      async_copy_b128(kl + 16, ks + 8);
      async_copy_b128(vl,      vs);
      async_copy_b128(vl + 16, vs + 8);
      asm volatile("s_wait_asynccnt 0x4" ::: "memory");  // tile `it` resident
    } else {
      asm volatile("s_wait_asynccnt 0x0" ::: "memory");
    }
    __syncthreads();                    // all waves' tile `it` DMA done

    const unsigned short* cK = (const unsigned short*)sK4[it & 1];
    const unsigned short* cV = (const unsigned short*)sV4[it & 1];

    // ---- scores: Q(16x64) x K^T, two 16-key tiles
    v8f s0 = {}, s1 = {};
    {
      const unsigned short* kp0 = cK + lo * 64 + 8 * hi;         // keys j32+lo
      const unsigned short* kp1 = cK + (16 + lo) * 64 + 8 * hi;  // keys j32+16+lo
      Frag32 kb0, kb1, kc0, kc1;
      kb0.u[0] = *(const uint4*)(kp0);      kb0.u[1] = *(const uint4*)(kp0 + 16);
      kb1.u[0] = *(const uint4*)(kp0 + 32); kb1.u[1] = *(const uint4*)(kp0 + 48);
      kc0.u[0] = *(const uint4*)(kp1);      kc0.u[1] = *(const uint4*)(kp1 + 16);
      kc1.u[0] = *(const uint4*)(kp1 + 32); kc1.u[1] = *(const uint4*)(kp1 + 48);
      s0 = WMMA_BF16(qa0.v, kb0.v, s0);
      s0 = WMMA_BF16(qa1.v, kb1.v, s0);
      s1 = WMMA_BF16(qa0.v, kc0.v, s1);
      s1 = WMMA_BF16(qa1.v, kc1.v, s1);
    }

    // ---- scale + mask (mask depends on key column only)
    int mv0 = mg[j32 + lo];
    int mv1 = mg[j32 + 16 + lo];
    float smax[8];
    for (int r = 0; r < 8; ++r) {
      float a = (mv0 == 0) ? -1e9f : s0[r] * 0.125f;
      float c = (mv1 == 0) ? -1e9f : s1[r] * 0.125f;
      s0[r] = a; s1[r] = c;
      smax[r] = fmaxf(a, c);
    }
    for (int r = 0; r < 8; ++r)
      for (int d = 1; d < 16; d <<= 1)
        smax[r] = fmaxf(smax[r], __shfl_xor(smax[r], d, 32));

    float corr[8];
    for (int r = 0; r < 8; ++r) {
      float nm = fmaxf(mr[r], smax[r]);
      corr[r] = __expf(mr[r] - nm);
      mr[r] = nm;
      lr[r] *= corr[r];
    }
    for (int r = 0; r < 8; ++r) {
      o0[r] *= corr[r]; o1[r] *= corr[r]; o2[r] *= corr[r]; o3[r] *= corr[r];
    }
    float rsum[8];
    for (int r = 0; r < 8; ++r) {
      float p0 = __expf(s0[r] - mr[r]);
      float p1 = __expf(s1[r] - mr[r]);
      s0[r] = p0; s1[r] = p1;
      rsum[r] = p0 + p1;
    }
    for (int r = 0; r < 8; ++r)
      for (int d = 1; d < 16; d <<= 1)
        rsum[r] += __shfl_xor(rsum[r], d, 32);
    for (int r = 0; r < 8; ++r) lr[r] += rsum[r];

    // ---- D-layout P -> per-wave LDS (16x32 row-major) -> reload in A-layout
    for (int r = 0; r < 8; ++r) {
      int m = r + 8 * hi;
      myp[m * 32 + lo]      = f2bf(s0[r]);
      myp[m * 32 + 16 + lo] = f2bf(s1[r]);
    }
    asm volatile("s_wait_dscnt 0" ::: "memory");   // wave-local store->load
    Frag32 pa;
    {
      const unsigned short* pp = myp + lo * 32 + 8 * hi;
      pa.u[0] = *(const uint4*)(pp);
      pa.u[1] = *(const uint4*)(pp + 16);
    }

    // ---- P(16x32) @ V(32x64): B-frags contiguous from d-major sV
    for (int dt = 0; dt < 4; ++dt) {
      const unsigned short* vp = cV + (dt * 16 + lo) * 32 + 8 * hi;
      Frag32 vb;
      vb.u[0] = *(const uint4*)(vp);
      vb.u[1] = *(const uint4*)(vp + 16);
      v8f* od = (dt == 0) ? &o0 : (dt == 1) ? &o1 : (dt == 2) ? &o2 : &o3;
      *od = WMMA_BF16(pa.v, vb.v, *od);
    }
    __syncthreads();                    // tile `it` buffers free for reuse
  }

  // ---- normalize + store to (B,S,1024) bf16, merged-head layout
  for (int r = 0; r < 8; ++r) {
    float inv = 1.0f / lr[r];
    int srow = q0 + r + 8 * hi;
    size_t base = ((size_t)b * 2048 + srow) * 1024 + h * 64 + lo;
    Ob[base]      = f2bf(o0[r] * inv);
    Ob[base + 16] = f2bf(o1[r] * inv);
    Ob[base + 32] = f2bf(o2[r] * inv);
    Ob[base + 48] = f2bf(o3[r] * inv);
  }
}

// ---------------------------------------------------------------- launcher
extern "C" void kernel_launch(void* const* d_in, const int* in_sizes, int n_in,
                              void* d_out, int out_size, void* d_ws, size_t ws_size,
                              hipStream_t stream) {
  const float* q    = (const float*)d_in[0];
  const float* k    = (const float*)d_in[1];
  const float* v    = (const float*)d_in[2];
  const int*   mask = (const int*)d_in[3];
  const float* Wq   = (const float*)d_in[4];
  const float* bq   = (const float*)d_in[5];
  const float* Wk   = (const float*)d_in[6];
  const float* bk   = (const float*)d_in[7];
  const float* Wv   = (const float*)d_in[8];
  const float* bv   = (const float*)d_in[9];
  const float* Wo   = (const float*)d_in[10];
  const float* bo   = (const float*)d_in[11];

  unsigned short* ws = (unsigned short*)d_ws;
  const size_t WN = 1024u * 1024u;        // weight elements
  const size_t XN = 2u * 2048u * 1024u;   // activation elements (B*S*D)

  unsigned short* Wq_b = ws;
  unsigned short* Wk_b = ws + WN;
  unsigned short* Wv_b = ws + 2 * WN;
  unsigned short* Wo_b = ws + 3 * WN;
  unsigned short* Xq_b = ws + 4 * WN;
  unsigned short* Xk_b = Xq_b + XN;
  unsigned short* Xv_b = Xk_b + XN;
  unsigned short* Qhp  = Xv_b + XN;       // (b,h,s,64) bf16
  unsigned short* Khp  = Qhp + XN;        // (b,h,s,64) bf16
  unsigned short* Vtp  = Khp + XN;        // (b,h,64,s) bf16 (transposed)
  unsigned short* Ab   = Vtp + XN;        // attn output, bf16 (B,S,D)

  cvt_f32_bf16<<<1024, 256, 0, stream>>>(Wq, Wq_b, (int)WN);
  cvt_f32_bf16<<<1024, 256, 0, stream>>>(Wk, Wk_b, (int)WN);
  cvt_f32_bf16<<<1024, 256, 0, stream>>>(Wv, Wv_b, (int)WN);
  cvt_f32_bf16<<<1024, 256, 0, stream>>>(Wo, Wo_b, (int)WN);
  cvt_f32_bf16<<<2048, 256, 0, stream>>>(q, Xq_b, (int)XN);
  cvt_f32_bf16<<<2048, 256, 0, stream>>>(k, Xk_b, (int)XN);
  cvt_f32_bf16<<<2048, 256, 0, stream>>>(v, Xv_b, (int)XN);

  // 128 x 32 = 4096 wave-tiles of 32x32, 8 waves per 256-thread block
  proj32<<<512, 256, 0, stream>>>(Xq_b, Wq_b, bq, Qhp, 0);
  proj32<<<512, 256, 0, stream>>>(Xk_b, Wk_b, bk, Khp, 0);
  proj32<<<512, 256, 0, stream>>>(Xv_b, Wv_b, bv, Vtp, 1);

  // 32 (b,h) x 128 q-tiles = 4096 waves, 4 waves per 128-thread block
  flash64<<<1024, 128, 0, stream>>>(Qhp, Khp, Vtp, mask, Ab);

  proj_out32<<<512, 256, 0, stream>>>(Ab, Wo_b, bo, (float*)d_out);
}
